// DH_54262616817679
// MI455X (gfx1250) — matrix operand order repaired and checked by reading, not compiled, not run on hardware
//
#include <hip/hip_runtime.h>
#include <hip/hip_bf16.h>
#include <math.h>

typedef __bf16 bf16_t;
typedef bf16_t v16bf __attribute__((ext_vector_type(16)));
typedef float  v8f   __attribute__((ext_vector_type(8)));

#define BN_EPS 1e-5f

__device__ __forceinline__ float sigmoidf_(float x) { return 1.0f / (1.0f + expf(-x)); }

// CDNA5 async copy: global -> LDS, 16B per lane, tracked by ASYNCcnt.
__device__ __forceinline__ void async_copy_b128(unsigned lds_addr, const void* gptr)
{
    asm volatile("global_load_async_to_lds_b128 %0, %1, off"
                 :: "v"(lds_addr), "v"(gptr)
                 : "memory");
}
__device__ __forceinline__ void wait_async0()
{
    asm volatile("s_wait_asynccnt 0x0" ::: "memory");
}
// low 32 bits of a generic pointer into LDS == wave-relative LDS byte address
__device__ __forceinline__ unsigned lds_addr_of(const void* p)
{
    return (unsigned)(unsigned long long)p;
}

// ---------------- weight packing: f32 (OIHW flat = [Cout][K]) -> bf16 [Cout][Kpad]
__global__ void pack_w_kernel(const float* __restrict__ w, bf16_t* __restrict__ wp,
                              int Cout, int K, int Kpad)
{
    long idx = (long)blockIdx.x * blockDim.x + threadIdx.x;
    long total = (long)Cout * Kpad;
    if (idx >= total) return;
    int o = (int)(idx / Kpad);
    int k = (int)(idx - (long)o * Kpad);
    float v = (k < K) ? w[(long)o * K + k] : 0.0f;
    wp[idx] = (bf16_t)v;
}

// ---------------- WMMA implicit-GEMM conv ----------------
// D[Cout][M] = Wp[Cout][Kpad] * X[Kpad][M] (+bias, opt ReLU, opt *scale)
// Macro tile: 128 (Cout) x 128 (pixels), 8 waves, 8 WMMA / wave / K-step,
// double-buffered LDS with async global->LDS DMA overlapped with WMMA.
// MODE_G3: X gathered on the fly (3x3, pad 1, stride 1), k = c*9 + (ky*3+kx)
// MODE_G1: X gathered on the fly (1x1), k = c
// MODE_COL: X async-DMA'd from prebuilt bf16 column buffer col[M][Kpad]
// src element (b,c,y,x) at src[b*sb + c*sc + y*W + x]  (NCHW or channel-major)
enum { MODE_G3 = 0, MODE_G1 = 1, MODE_COL = 2 };

template<int MODE, bool RELU, bool NCHW>
__global__ __launch_bounds__(256)
void gemm_conv_wmma(const bf16_t* __restrict__ Wp,
                    const float* __restrict__ bias,
                    const float* __restrict__ src,
                    const bf16_t* __restrict__ col,
                    float* __restrict__ out,
                    int Cout, int Kpad, int Mtot,
                    int Cin, int H, int W,
                    long sb, long sc,
                    int HWo, float oscale)
{
    __shared__ __align__(16) bf16_t lsA[2][128 * 32];   // weights   [128 rows][32 k]
    __shared__ __align__(16) bf16_t lsB[2][128 * 32];   // activ col [128 pix ][32 k]
    __shared__ int pb[128], py[128], px[128];

    const int tid = threadIdx.x;
    const int m0  = blockIdx.x * 128;
    const int n0  = blockIdx.y * 128;
    const int HW  = H * W;

    if (tid < 128) {
        int m = m0 + tid;
        if (m < Mtot) {
            int b  = m / HW;
            int hw = m - b * HW;
            int yy = hw / W;
            pb[tid] = b; py[tid] = yy; px[tid] = hw - yy * W;
        } else { pb[tid] = -1; py[tid] = 0; px[tid] = 0; }
    }
    __syncthreads();

    const int wv   = tid >> 5;
    const int lane = tid & 31;
    const int tnb  = (wv & 3) << 1;     // 2 row-tiles of 16 per wave (8 total)
    const int tmb  = (wv >> 2) << 2;    // 4 col-tiles of 16 per wave (8 total)

    v8f acc0[4] = {{}, {}, {}, {}};
    v8f acc1[4] = {{}, {}, {}, {}};

    const uint4 z4 = make_uint4(0u, 0u, 0u, 0u);

    // ---- staging helpers (issue async DMA / gather into LDS buffer `buf`) ----
    auto stageA = [&](int buf, int kk) {
        #pragma unroll
        for (int i = 0; i < 2; ++i) {
            int cid  = tid * 2 + i;          // 0..511 chunks of 8 bf16
            int row  = cid >> 2;             // 0..127
            int koff = (cid & 3) << 3;
            int n = n0 + row;
            bf16_t* dst = &lsA[buf][row * 32 + koff];
            if (n < Cout)
                async_copy_b128(lds_addr_of(dst), &Wp[(long)n * Kpad + kk + koff]);
            else
                *reinterpret_cast<uint4*>(dst) = z4;
        }
    };
    auto stageB = [&](int buf, int kk) {
        if (MODE == MODE_COL) {
            #pragma unroll
            for (int i = 0; i < 2; ++i) {
                int cid  = tid * 2 + i;
                int p    = cid >> 2;
                int koff = (cid & 3) << 3;
                bf16_t* dst = &lsB[buf][p * 32 + koff];
                if (m0 + p < Mtot)
                    async_copy_b128(lds_addr_of(dst), &col[(long)(m0 + p) * Kpad + kk + koff]);
                else
                    *reinterpret_cast<uint4*>(dst) = z4;
            }
        } else {
            #pragma unroll
            for (int i = 0; i < 16; ++i) {
                int e = i * 256 + tid;       // 0..4095
                int k = e >> 7;              // 0..31
                int p = e & 127;
                float v = 0.0f;
                int b = pb[p];
                if (b >= 0) {
                    int kg = kk + k;
                    if (MODE == MODE_G3) {
                        int c  = kg / 9;
                        int t  = kg - 9 * c;
                        int t3 = t / 3;
                        int iy = py[p] + t3 - 1;
                        int ix = px[p] + (t - 3 * t3) - 1;
                        if (c < Cin && iy >= 0 && iy < H && ix >= 0 && ix < W)
                            v = src[(long)b * sb + (long)c * sc + (long)iy * W + ix];
                    } else {
                        if (kg < Cin)
                            v = src[(long)b * sb + (long)kg * sc + (long)py[p] * W + px[p]];
                    }
                }
                lsB[buf][p * 32 + k] = (bf16_t)v;
            }
        }
    };

    // ---- prologue: fill buffer 0 ----
    stageA(0, 0);
    stageB(0, 0);
    wait_async0();
    __syncthreads();

    // ---- K loop: stage next slab async while computing current ----
    for (int kk = 0; kk < Kpad; kk += 32) {
        const int cur = (kk >> 5) & 1;
        const int nxt = cur ^ 1;
        if (kk + 32 < Kpad) {
            stageA(nxt, kk + 32);
            stageB(nxt, kk + 32);
        }

        union FA { uint4 u[2]; v16bf v; } fa0, fa1;
        {
            const bf16_t* A = lsA[cur];
            int r0 = (tnb + 0) * 16 + (lane & 15);
            int r1 = (tnb + 1) * 16 + (lane & 15);
            int kf = (lane >> 4) << 3;                // 0 or 8
            fa0.u[0] = *reinterpret_cast<const uint4*>(&A[r0 * 32 + kf]);
            fa0.u[1] = *reinterpret_cast<const uint4*>(&A[r0 * 32 + kf + 16]);
            fa1.u[0] = *reinterpret_cast<const uint4*>(&A[r1 * 32 + kf]);
            fa1.u[1] = *reinterpret_cast<const uint4*>(&A[r1 * 32 + kf + 16]);
        }
        {
            const bf16_t* Bp = lsB[cur];
            const int kb = (lane >> 4) << 3;          // 0 or 8
            #pragma unroll
            for (int j = 0; j < 4; ++j) {
                union FB { uint4 u[2]; v16bf v; } fb;
                int p = (tmb + j) * 16 + (lane & 15);
                fb.u[0] = *reinterpret_cast<const uint4*>(&Bp[p * 32 + kb]);
                fb.u[1] = *reinterpret_cast<const uint4*>(&Bp[p * 32 + kb + 16]);
                acc0[j] = __builtin_amdgcn_wmma_f32_16x16x32_bf16(false, fa0.v, false, fb.v,
                                                                  (short)0, acc0[j], false, false);
                acc1[j] = __builtin_amdgcn_wmma_f32_16x16x32_bf16(false, fa1.v, false, fb.v,
                                                                  (short)0, acc1[j], false, false);
            }
        }
        wait_async0();
        __syncthreads();
    }

    // ---- epilogue: bias (+ReLU) (*scale), scatter per D-layout ----
    const int pcol = lane & 15;
    #pragma unroll
    for (int i = 0; i < 2; ++i) {
        const int nbase = n0 + (tnb + i) * 16 + ((lane >> 4) << 3);
        #pragma unroll
        for (int j = 0; j < 4; ++j) {
            int m = m0 + (tmb + j) * 16 + pcol;
            if (m >= Mtot) continue;
            #pragma unroll
            for (int v = 0; v < 8; ++v) {
                int n = nbase + v;
                if (n >= Cout) continue;
                float r = (i == 0 ? acc0[j][v] : acc1[j][v]) + bias[n];
                if (RELU) r = fmaxf(r, 0.0f);
                r *= oscale;
                long addr;
                if (NCHW) {
                    int b  = m / HWo;
                    int hw = m - b * HWo;
                    addr = ((long)b * Cout + n) * HWo + hw;
                } else {
                    addr = (long)n * Mtot + m;
                }
                out[addr] = r;
            }
        }
    }
}

// ---------------- DCNv2 modulated bilinear im2col -> bf16 columns [M][Kpad]
// om is [27][B][H][W] channel-major: ch 0-8 off_y, 9-17 off_x, 18-26 mask logits
__global__ __launch_bounds__(256)
void deform_im2col_kernel(const float* __restrict__ src, long sb, long sc,
                          const float* __restrict__ om,
                          bf16_t* __restrict__ col,
                          int Cin, int B, int H, int W, int Kpad)
{
    int HW = H * W;
    long total = (long)B * HW * 9;
    long idx = (long)blockIdx.x * blockDim.x + threadIdx.x;
    if (idx >= total) return;
    int  t  = (int)(idx % 9);
    long m  = idx / 9;
    int  b  = (int)(m / HW);
    int  hw = (int)(m - (long)b * HW);
    int  y  = hw / W, x = hw - (hw / W) * W;
    long ob  = (long)b * HW + hw;
    long chs = (long)B * HW;
    float offy = om[(long)t * chs + ob];
    float offx = om[(long)(9 + t) * chs + ob];
    float mk   = sigmoidf_(om[(long)(18 + t) * chs + ob]);
    float gy = offy + (float)y + (float)(t / 3 - 1);
    float gx = offx + (float)x + (float)(t % 3 - 1);
    float y0f = floorf(gy), x0f = floorf(gx);
    int y0 = (int)y0f, x0 = (int)x0f;
    float ty = gy - y0f, tx = gx - x0f;
    int y1 = y0 + 1, x1 = x0 + 1;
    bool vy0 = (y0 >= 0) && (y0 < H), vy1 = (y1 >= 0) && (y1 < H);
    bool vx0 = (x0 >= 0) && (x0 < W), vx1 = (x1 >= 0) && (x1 < W);
    float w00 = (1 - ty) * (1 - tx) * mk, w01 = (1 - ty) * tx * mk;
    float w10 = ty * (1 - tx) * mk,       w11 = ty * tx * mk;
    const float* p0 = src + (long)b * sb;
    bf16_t* cp = col + m * Kpad + t;
    for (int c = 0; c < Cin; ++c) {
        const float* pc = p0 + (long)c * sc;
        float a = 0.0f;
        if (vy0 && vx0) a += w00 * pc[(long)y0 * W + x0];
        if (vy0 && vx1) a += w01 * pc[(long)y0 * W + x1];
        if (vy1 && vx0) a += w10 * pc[(long)y1 * W + x0];
        if (vy1 && vx1) a += w11 * pc[(long)y1 * W + x1];
        cp[(long)c * 9] = (bf16_t)a;
    }
}

// ---------------- per-channel batch stats over channel-major [C][M]
__global__ __launch_bounds__(256)
void bn_stats_kernel(const float* __restrict__ x, long M,
                     float* __restrict__ mean, float* __restrict__ istd)
{
    __shared__ float rs[256], rq[256];
    int c = blockIdx.x;
    const float* p = x + (long)c * M;
    float s = 0.0f, q = 0.0f;
    for (long i = threadIdx.x; i < M; i += 256) { float v = p[i]; s += v; q += v * v; }
    rs[threadIdx.x] = s; rq[threadIdx.x] = q;
    __syncthreads();
    for (int st = 128; st > 0; st >>= 1) {
        if (threadIdx.x < st) {
            rs[threadIdx.x] += rs[threadIdx.x + st];
            rq[threadIdx.x] += rq[threadIdx.x + st];
        }
        __syncthreads();
    }
    if (threadIdx.x == 0) {
        float mu  = rs[0] / (float)M;
        float var = rq[0] / (float)M - mu * mu;
        mean[c] = mu;
        istd[c] = rsqrtf(var + BN_EPS);
    }
}

// ---------------- fused BN(train) + ReLU + 2x bilinear (align_corners) + skip add
// conv: [C][B][H][W] channel-major; base: NCHW [B][C][2H][2W]; out: [C][B][2H][2W]
__global__ __launch_bounds__(256)
void bn_up_add_kernel(const float* __restrict__ conv,
                      const float* __restrict__ mean, const float* __restrict__ istd,
                      const float* __restrict__ g, const float* __restrict__ be,
                      const float* __restrict__ base,
                      float* __restrict__ out,
                      int C, int B, int H, int W)
{
    int W2 = 2 * W, H2 = 2 * H;
    long total = (long)C * B * H2 * W2;
    long idx = (long)blockIdx.x * blockDim.x + threadIdx.x;
    if (idx >= total) return;
    int X = (int)(idx % W2); long r = idx / W2;
    int Y = (int)(r % H2);  r /= H2;
    int b = (int)(r % B);   int c = (int)(r / B);
    float sy = (float)(H - 1) / (float)(H2 - 1);
    float sx = (float)(W - 1) / (float)(W2 - 1);
    float fy = Y * sy, fx = X * sx;
    float y0f = floorf(fy), x0f = floorf(fx);
    int y0 = (int)y0f, x0 = (int)x0f;
    int y1 = (y0 + 1 < H) ? y0 + 1 : H - 1;
    int x1 = (x0 + 1 < W) ? x0 + 1 : W - 1;
    float ty = fy - y0f, tx = fx - x0f;
    const float* cp = conv + ((long)c * B + b) * H * W;
    float mu = mean[c], is = istd[c], ga = g[c], bb = be[c];
    auto val = [&](int yy, int xx) {
        float v = cp[(long)yy * W + xx];
        v = (v - mu) * is * ga + bb;
        return fmaxf(v, 0.0f);
    };
    float up = val(y0, x0) * (1 - ty) * (1 - tx) + val(y0, x1) * (1 - ty) * tx
             + val(y1, x0) * ty * (1 - tx)       + val(y1, x1) * ty * tx;
    float bv = base[((long)b * C + c) * (long)H2 * W2 + (long)Y * W2 + X];
    out[((long)c * B + b) * (long)H2 * W2 + (long)Y * W2 + X] = bv + up;
}

// ================= host side =================
extern "C" void kernel_launch(void* const* d_in, const int* in_sizes, int n_in,
                              void* d_out, int out_size, void* d_ws, size_t ws_size,
                              hipStream_t stream)
{
    (void)in_sizes; (void)n_in; (void)out_size; (void)ws_size;
    const float* x1    = (const float*)d_in[0];
    const float* x2    = (const float*)d_in[1];
    const float* x3    = (const float*)d_in[2];
    const float* d1_ow = (const float*)d_in[3];
    const float* d1_ob = (const float*)d_in[4];
    const float* d1_w  = (const float*)d_in[5];
    const float* d1_b  = (const float*)d_in[6];
    const float* d1_g  = (const float*)d_in[7];
    const float* d1_be = (const float*)d_in[8];
    const float* d2_ow = (const float*)d_in[9];
    const float* d2_ob = (const float*)d_in[10];
    const float* d2_w  = (const float*)d_in[11];
    const float* d2_b  = (const float*)d_in[12];
    const float* d2_g  = (const float*)d_in[13];
    const float* d2_be = (const float*)d_in[14];
    const float* hm_w1 = (const float*)d_in[15];
    const float* hm_b1 = (const float*)d_in[16];
    const float* hm_w2 = (const float*)d_in[17];
    const float* hm_b2 = (const float*)d_in[18];
    const float* hm_w3 = (const float*)d_in[19];
    const float* hm_b3 = (const float*)d_in[20];
    const float* wh_w1 = (const float*)d_in[21];
    const float* wh_b1 = (const float*)d_in[22];
    const float* wh_w2 = (const float*)d_in[23];
    const float* wh_b2 = (const float*)d_in[24];
    const float* wh_w3 = (const float*)d_in[25];
    const float* wh_b3 = (const float*)d_in[26];
    float* out = (float*)d_out;

    const int B = 8;
    const int M1 = B * 32 * 32;       // 8192
    const int M2 = B * 64 * 64;       // 32768
    const int M3 = B * 128 * 128;     // 131072

    size_t off = 0;
    auto alloc = [&](size_t bytes) -> void* {
        void* p = (void*)((char*)d_ws + off);
        off += (bytes + 255) & ~(size_t)255;
        return p;
    };
    bf16_t* wp_d1ow = (bf16_t*)alloc((size_t)27  * 4608 * 2);
    bf16_t* wp_d1   = (bf16_t*)alloc((size_t)256 * 4608 * 2);
    bf16_t* wp_d2ow = (bf16_t*)alloc((size_t)27  * 2304 * 2);
    bf16_t* wp_d2   = (bf16_t*)alloc((size_t)128 * 2304 * 2);
    bf16_t* wp_hm1  = (bf16_t*)alloc((size_t)128 * 1152 * 2);
    bf16_t* wp_hm2  = (bf16_t*)alloc((size_t)128 * 1152 * 2);
    bf16_t* wp_hm3  = (bf16_t*)alloc((size_t)80  * 128  * 2);
    bf16_t* wp_wh1  = (bf16_t*)alloc((size_t)128 * 1152 * 2);
    bf16_t* wp_wh2  = (bf16_t*)alloc((size_t)64  * 1152 * 2);
    bf16_t* wp_wh3  = (bf16_t*)alloc((size_t)4   * 64   * 2);
    float*  om      = (float*)alloc((size_t)27 * M2 * 4);            // reused lvl1/lvl2
    bf16_t* colb    = (bf16_t*)alloc((size_t)M2 * 2304 * 2);         // reused (lvl2 is max)
    float*  dconv   = (float*)alloc((size_t)128 * M2 * 4);           // reused d1out/d2out
    float*  meanb   = (float*)alloc((size_t)256 * 4);
    float*  istdb   = (float*)alloc((size_t)256 * 4);
    float*  x2new   = (float*)alloc((size_t)256 * M2 * 4);           // [C][B][64][64]
    float*  x3new   = (float*)alloc((size_t)128 * M3 * 4);           // [C][B][128][128]
    float*  bufA    = (float*)alloc((size_t)128 * M3 * 4);
    float*  bufB    = (float*)alloc((size_t)128 * M3 * 4);

    auto pack = [&](const float* w, bf16_t* wp, int Cout, int K) {
        long total = (long)Cout * K;
        pack_w_kernel<<<dim3((unsigned)((total + 255) / 256)), 256, 0, stream>>>(w, wp, Cout, K, K);
    };
    pack(d1_ow, wp_d1ow, 27, 4608);  pack(d1_w, wp_d1, 256, 4608);
    pack(d2_ow, wp_d2ow, 27, 2304);  pack(d2_w, wp_d2, 128, 2304);
    pack(hm_w1, wp_hm1, 128, 1152);  pack(hm_w2, wp_hm2, 128, 1152);  pack(hm_w3, wp_hm3, 80, 128);
    pack(wh_w1, wp_wh1, 128, 1152);  pack(wh_w2, wp_wh2, 64, 1152);   pack(wh_w3, wp_wh3, 4, 64);

    auto gdim = [](int M, int N) {
        return dim3((unsigned)((M + 127) / 128), (unsigned)((N + 127) / 128));
    };

    // ======== level 1: x1 (NCHW 8,512,32,32) ========
    const long x1_sb = 512L * 1024, x1_sc = 1024;
    gemm_conv_wmma<MODE_G3, false, false><<<gdim(M1, 27), 256, 0, stream>>>(
        wp_d1ow, d1_ob, x1, nullptr, om, 27, 4608, M1, 512, 32, 32, x1_sb, x1_sc, 0, 1.0f);
    {
        long total = (long)M1 * 9;
        deform_im2col_kernel<<<dim3((unsigned)((total + 255) / 256)), 256, 0, stream>>>(
            x1, x1_sb, x1_sc, om, colb, 512, B, 32, 32, 4608);
    }
    gemm_conv_wmma<MODE_COL, false, false><<<gdim(M1, 256), 256, 0, stream>>>(
        wp_d1, d1_b, nullptr, colb, dconv, 256, 4608, M1, 512, 32, 32, 0, 0, 0, 1.0f);
    bn_stats_kernel<<<256, 256, 0, stream>>>(dconv, (long)M1, meanb, istdb);
    {
        long total = (long)256 * M2;
        bn_up_add_kernel<<<dim3((unsigned)((total + 255) / 256)), 256, 0, stream>>>(
            dconv, meanb, istdb, d1_g, d1_be, x2, x2new, 256, B, 32, 32);
    }

    // ======== level 2: x2new ([C][B][64][64], sb=HW, sc=B*HW) ========
    const long x2_sb = 4096, x2_sc = (long)B * 4096;
    gemm_conv_wmma<MODE_G3, false, false><<<gdim(M2, 27), 256, 0, stream>>>(
        wp_d2ow, d2_ob, x2new, nullptr, om, 27, 2304, M2, 256, 64, 64, x2_sb, x2_sc, 0, 1.0f);
    {
        long total = (long)M2 * 9;
        deform_im2col_kernel<<<dim3((unsigned)((total + 255) / 256)), 256, 0, stream>>>(
            x2new, x2_sb, x2_sc, om, colb, 256, B, 64, 64, 2304);
    }
    gemm_conv_wmma<MODE_COL, false, false><<<gdim(M2, 128), 256, 0, stream>>>(
        wp_d2, d2_b, nullptr, colb, dconv, 128, 2304, M2, 256, 64, 64, 0, 0, 0, 1.0f);
    bn_stats_kernel<<<128, 256, 0, stream>>>(dconv, (long)M2, meanb, istdb);
    {
        long total = (long)128 * M3;
        bn_up_add_kernel<<<dim3((unsigned)((total + 255) / 256)), 256, 0, stream>>>(
            dconv, meanb, istdb, d2_g, d2_be, x3, x3new, 128, B, 64, 64);
    }

    // ======== heads: x3new ([C][B][128][128]) ========
    const long x3_sb = 16384, x3_sc = (long)B * 16384;
    // hm head
    gemm_conv_wmma<MODE_G3, true, false><<<gdim(M3, 128), 256, 0, stream>>>(
        wp_hm1, hm_b1, x3new, nullptr, bufA, 128, 1152, M3, 128, 128, 128, x3_sb, x3_sc, 0, 1.0f);
    gemm_conv_wmma<MODE_G3, true, false><<<gdim(M3, 128), 256, 0, stream>>>(
        wp_hm2, hm_b2, bufA, nullptr, bufB, 128, 1152, M3, 128, 128, 128, x3_sb, x3_sc, 0, 1.0f);
    gemm_conv_wmma<MODE_G1, false, true><<<gdim(M3, 80), 256, 0, stream>>>(
        wp_hm3, hm_b3, bufB, nullptr, out, 80, 128, M3, 128, 128, 128, x3_sb, x3_sc, 16384, 1.0f);
    // wh head
    gemm_conv_wmma<MODE_G3, true, false><<<gdim(M3, 128), 256, 0, stream>>>(
        wp_wh1, wh_b1, x3new, nullptr, bufA, 128, 1152, M3, 128, 128, 128, x3_sb, x3_sc, 0, 1.0f);
    gemm_conv_wmma<MODE_G3, true, false><<<gdim(M3, 64), 256, 0, stream>>>(
        wp_wh2, wh_b2, bufA, nullptr, bufB, 64, 1152, M3, 128, 128, 128, x3_sb, x3_sc, 0, 1.0f);
    gemm_conv_wmma<MODE_G1, true, true><<<gdim(M3, 4), 256, 0, stream>>>(
        wp_wh3, wh_b3, bufB, nullptr, out + (long)B * 80 * 16384, 4, 64, M3, 64, 128, 128,
        x3_sb, x3_sc, 16384, 16.0f);
}
